// GPT_21079699489192
// MI455X (gfx1250) — matrix operand order repaired and checked by reading, not compiled, not run on hardware
//
#include <hip/hip_runtime.h>
#include <math.h>

// ---------------- Model dimensions (GPT-2 small, 6 layers) ----------------
#define NV     50257
#define EMB    768
#define NH     12
#define HDIM   64
#define NL     6
#define SEQ    1024
#define NBATCH 2
#define NROWS  (NBATCH*SEQ)   // 2048 token rows

// ---------------- vector types ----------------
typedef __attribute__((ext_vector_type(16))) __bf16 bf16x16;
typedef __attribute__((ext_vector_type(8)))  __bf16 bf16x8;
typedef __attribute__((ext_vector_type(4)))  __bf16 bf16x4;
typedef __attribute__((ext_vector_type(8)))  float  f32x8;
typedef __attribute__((ext_vector_type(4)))  float  f32x4;
typedef int v4i_gcc __attribute__((vector_size(16)));   // matches async builtin param

// gfx1250 async global->LDS copy (ASYNCcnt-tracked), guarded by __has_builtin
#if __has_builtin(__builtin_amdgcn_global_load_async_to_lds_b128) && \
    __has_builtin(__builtin_amdgcn_s_wait_asynccnt)
#define USE_ASYNC_LDS 1
#else
#define USE_ASYNC_LDS 0
#endif

// fp32 -> bf16 round-to-nearest-even (bit trick)
__device__ __forceinline__ __bf16 f2bf(float f) {
    union { float f; unsigned u; } a; a.f = f;
    unsigned r = a.u + 0x7FFFu + ((a.u >> 16) & 1u);
    unsigned short h = (unsigned short)(r >> 16);
    __bf16 o;
    __builtin_memcpy(&o, &h, 2);
    return o;
}

// pack two floats into a dword of two bf16 (lo = a, hi = b)
__device__ __forceinline__ unsigned pack2bf(float a, float b) {
    union { __bf16 h[2]; unsigned u; } x;
    x.h[0] = f2bf(a); x.h[1] = f2bf(b);
    return x.u;
}

// 16B global->LDS copy: async DMA when available, else direct vector copy
__device__ __forceinline__ void copyA16(const __bf16* gsrc, __bf16* ldst) {
#if USE_ASYNC_LDS
    __builtin_amdgcn_global_load_async_to_lds_b128(
        (__attribute__((address_space(1))) v4i_gcc*)(gsrc),
        (__attribute__((address_space(3))) v4i_gcc*)(ldst), 0, 0);
#else
    *(bf16x8*)ldst = *(const bf16x8*)gsrc;
#endif
}

__device__ __forceinline__ float gelu_tanh(float x) {
    float x3 = x * x * x;
    return 0.5f * x * (1.0f + tanhf(0.7978845608028654f * (x + 0.044715f * x3)));
}

// ---------------- token + positional embedding ----------------
__global__ __launch_bounds__(256)
void embed_kernel(const int* __restrict__ x0, const float* __restrict__ wte,
                  const float* __restrict__ wpe, float* __restrict__ xres) {
    int row = blockIdx.x;          // b*SEQ + t
    int t   = row % SEQ;
    int tok = x0[row];
    const f32x4* we = (const f32x4*)(wte + (size_t)tok * EMB);
    const f32x4* pe = (const f32x4*)(wpe + (size_t)t * EMB);
    f32x4*       xo = (f32x4*)(xres + (size_t)row * EMB);
    for (int c = threadIdx.x; c < EMB / 4; c += 256)
        xo[c] = we[c] + pe[c];
}

// ---------------- LayerNorm (fp32 in, bf16 out) ----------------
__global__ __launch_bounds__(256)
void ln_kernel(const float* __restrict__ x, const float* __restrict__ w,
               const float* __restrict__ b, __bf16* __restrict__ out) {
    __shared__ float s1[256], s2[256];
    int row = blockIdx.x, tid = threadIdx.x;
    const f32x4* xr = (const f32x4*)(x + (size_t)row * EMB);
    float a = 0.f, q = 0.f;
    for (int c = tid; c < EMB / 4; c += 256) {
        f32x4 v = xr[c];
        #pragma unroll
        for (int j = 0; j < 4; j++) { a += v[j]; q += v[j] * v[j]; }
    }
    s1[tid] = a; s2[tid] = q; __syncthreads();
    for (int off = 128; off; off >>= 1) {
        if (tid < off) { s1[tid] += s1[tid + off]; s2[tid] += s2[tid + off]; }
        __syncthreads();
    }
    float mean = s1[0] * (1.0f / EMB);
    float var  = s2[0] * (1.0f / EMB) - mean * mean;
    float rstd = rsqrtf(var + 1e-6f);
    const f32x4* wv = (const f32x4*)w;
    const f32x4* bv = (const f32x4*)b;
    bf16x4* ov = (bf16x4*)(out + (size_t)row * EMB);
    for (int c = tid; c < EMB / 4; c += 256) {
        f32x4 v = xr[c], wc = wv[c], bc = bv[c];
        bf16x4 o;
        #pragma unroll
        for (int j = 0; j < 4; j++) o[j] = f2bf((v[j] - mean) * rstd * wc[j] + bc[j]);
        ov[c] = o;
    }
}

// ---------------- fp32 -> bf16 elementwise (vectorized x4) ----------------
__global__ __launch_bounds__(256)
void cvt_bf16_kernel(const float* __restrict__ in, __bf16* __restrict__ out, int n4) {
    int i = blockIdx.x * 256 + threadIdx.x;
    if (i < n4) {
        f32x4 v = ((const f32x4*)in)[i];
        bf16x4 o;
        #pragma unroll
        for (int j = 0; j < 4; j++) o[j] = f2bf(v[j]);
        ((bf16x4*)out)[i] = o;
    }
}

// ---------------- WMMA bf16 GEMM: C[M,N] = A[M,K] @ B[K,N] (+res, +gelu) ----
// A: bf16 row-major. B: fp32; bT==0 -> [K,N] row-major (N % 64 == 0 required),
// bT==1 -> [N,K] row-major (logits; N edge guarded).
// Block = 128 threads = 4 waves; tile 128x64, K-step 32. Each wave owns a
// 32x64 strip: 2 A fragments x 4 B fragments -> 8 v_wmma per K-step.
// Double-buffered LDS; A tile via async DMA issued one step ahead so the
// tensor-path copy overlaps WMMA of the current step; B fp32 prefetched to
// registers one step ahead and converted to bf16 while the DMA flies.
__global__ __launch_bounds__(128)
void gemm_bf16_kernel(const __bf16* __restrict__ A, const float* __restrict__ Bm,
                      float* __restrict__ C, const float* __restrict__ res,
                      int N, int K, int bT, int doGelu) {
    __shared__ __align__(64) __bf16 As[2][128 * 32];   // [m][k]
    __shared__ __align__(64) __bf16 Bs[2][64 * 32];    // [n][k] transposed
    const int tid  = threadIdx.x;
    const int lane = tid & 31;
    const int wave = tid >> 5;
    const int hi   = lane >> 4;     // half-wave select
    const int lmod = lane & 15;
    const int m0   = blockIdx.y * 128;
    const int n0   = blockIdx.x * 64;

    f32x8 acc[2][4] = {};
    f32x4 pa[2], pb[2];             // B prefetch registers

    // ---- stage-A issue: 128x32 bf16 = 512 x 16B chunks, 4 per thread ----
    auto issueA = [&](int k0, int buf) {
        #pragma unroll
        for (int i = 0; i < 4; i++) {
            int c  = tid + 128 * i;           // 0..511
            int m  = c >> 2;                  // 0..127
            int ko = (c & 3) * 8;             // 0,8,16,24
            copyA16(A + (size_t)(m0 + m) * K + (k0 + ko), &As[buf][m * 32 + ko]);
        }
    };
    // ---- B prefetch (global fp32 -> registers) ----
    auto loadB = [&](int k0) {
        if (!bT) {
            #pragma unroll
            for (int i = 0; i < 2; i++) {
                int u  = tid + 128 * i;       // 0..255
                int k  = (u >> 4) * 2;        // 0,2,..,30
                int n4 = (u & 15) * 4;        // 0,4,..,60
                const float* r0 = Bm + (size_t)(k0 + k) * N + (n0 + n4);
                pa[i] = *(const f32x4*)r0;
                pb[i] = *(const f32x4*)(r0 + N);
            }
        } else {
            #pragma unroll
            for (int i = 0; i < 2; i++) {
                int u  = tid + 128 * i;       // 0..255
                int n  = u >> 2;              // 0..63
                int k  = (u & 3) * 8;         // 0,8,16,24
                int gn = n0 + n;
                f32x4 va = {}, vb = {};
                if (gn < N) {
                    const float* r0 = Bm + (size_t)gn * K + (k0 + k);
                    va = *(const f32x4*)r0;
                    vb = *(const f32x4*)(r0 + 4);
                }
                pa[i] = va; pb[i] = vb;
            }
        }
    };
    // ---- B store (registers -> bf16 LDS, [n][k] layout) ----
    auto storeB = [&](int buf) {
        if (!bT) {
            #pragma unroll
            for (int i = 0; i < 2; i++) {
                int u  = tid + 128 * i;
                int k  = (u >> 4) * 2;
                int n4 = (u & 15) * 4;
                #pragma unroll
                for (int j = 0; j < 4; j++)
                    *(unsigned*)(&Bs[buf][(n4 + j) * 32 + k]) = pack2bf(pa[i][j], pb[i][j]);
            }
        } else {
            #pragma unroll
            for (int i = 0; i < 2; i++) {
                int u = tid + 128 * i;
                int n = u >> 2;
                int k = (u & 3) * 8;
                bf16x8 pk;
                #pragma unroll
                for (int j = 0; j < 4; j++) { pk[j] = f2bf(pa[i][j]); pk[4 + j] = f2bf(pb[i][j]); }
                *(bf16x8*)(&Bs[buf][n * 32 + k]) = pk;
            }
        }
    };
    // ---- A fragment from LDS; contiguous k order (k = 16*hi + s), matching B ----
    auto loadAfrag = [&](const __bf16* rowbase) -> bf16x16 {
        bf16x8 c0 = *(const bf16x8*)(rowbase + hi * 16);
        bf16x8 c1 = *(const bf16x8*)(rowbase + hi * 16 + 8);
        bf16x16 af;
        #pragma unroll
        for (int j = 0; j < 8; j++) { af[j] = c0[j]; af[8 + j] = c1[j]; }
        return af;
    };

    // ---- software-pipelined main loop ----
    int buf = 0;
    issueA(0, 0);
    loadB(0);
    for (int k0 = 0; k0 < K; k0 += 32) {
        const bool more = (k0 + 32) < K;
        if (more) issueA(k0 + 32, buf ^ 1);   // DMA next tile (overlaps this step's WMMA)
        storeB(buf);                          // convert + store this step's B
        if (more) loadB(k0 + 32);             // prefetch next step's B fp32
#if USE_ASYNC_LDS
        if (more) __builtin_amdgcn_s_wait_asynccnt(4);   // drain tile k0, keep k0+32 in flight
        else      __builtin_amdgcn_s_wait_asynccnt(0);
#endif
        __syncthreads();

        bf16x16 af0 = loadAfrag(&As[buf][(wave * 32 + lmod) * 32]);
        bf16x16 af1 = loadAfrag(&As[buf][(wave * 32 + 16 + lmod) * 32]);
        #pragma unroll
        for (int nt = 0; nt < 4; nt++) {
            bf16x16 bfr = *(const bf16x16*)(&Bs[buf][(nt * 16 + lmod) * 32 + hi * 16]);
            acc[0][nt] = __builtin_amdgcn_wmma_f32_16x16x32_bf16(
                false, af0, false, bfr, (short)0, acc[0][nt], false, false);
            acc[1][nt] = __builtin_amdgcn_wmma_f32_16x16x32_bf16(
                false, af1, false, bfr, (short)0, acc[1][nt], false, false);
        }
        __syncthreads();   // all waves done with buf before it is re-targeted
        buf ^= 1;
    }

    // epilogue: C layout — VGPR r, lanes 0-15: M = r, lanes 16-31: M = 8 + r
    #pragma unroll
    for (int half = 0; half < 2; half++) {
        #pragma unroll
        for (int nt = 0; nt < 4; nt++) {
            int gn = n0 + nt * 16 + lmod;
            if (gn < N) {
                #pragma unroll
                for (int r = 0; r < 8; r++) {
                    int gm = m0 + wave * 32 + half * 16 + hi * 8 + r;
                    float v = acc[half][nt][r];
                    if (doGelu) v = gelu_tanh(v);
                    size_t idx = (size_t)gm * N + gn;
                    if (res) v += res[idx];
                    C[idx] = v;
                }
            }
        }
    }
}

// ---------------- causal attention, online softmax, wave-per-query ----------
// qkv: [NROWS, 3*EMB] fp32 (q | k | v). y: [NROWS, EMB] fp32.
__global__ __launch_bounds__(256)
void attn_kernel(const float* __restrict__ qkv, float* __restrict__ y) {
    __shared__ float qsh[8][64];
    int tid  = threadIdx.x, lane = tid & 31, wave = tid >> 5;
    int gq = blockIdx.x * 8 + wave;           // global (b,h,q) index
    int q  = gq % SEQ;
    int h  = (gq / SEQ) % NH;
    int b  = gq / (SEQ * NH);
    size_t rq = (size_t)(b * SEQ + q) * (3 * EMB);
    qsh[wave][lane]      = qkv[rq + h * HDIM + lane];
    qsh[wave][lane + 32] = qkv[rq + h * HDIM + lane + 32];
    __syncthreads();

    const float scale = 0.125f;               // 1/sqrt(64)
    float m = -1e30f, l = 0.f, acc0 = 0.f, acc1 = 0.f;

    for (int s0 = 0; s0 <= q; s0 += 32) {
        int s = s0 + lane;
        float sc = -1e30f;
        if (s <= q) {
            const float* kr = qkv + (size_t)(b * SEQ + s) * (3 * EMB) + EMB + h * HDIM;
            float d = 0.f;
            #pragma unroll
            for (int dd = 0; dd < 64; dd++) d += qsh[wave][dd] * kr[dd];
            sc = d * scale;
        }
        float tmax = sc;
        for (int off = 16; off; off >>= 1) tmax = fmaxf(tmax, __shfl_xor(tmax, off, 32));
        float mnew = fmaxf(m, tmax);
        float corr = __expf(m - mnew);
        float p = (s <= q) ? __expf(sc - mnew) : 0.f;
        float tsum = p;
        for (int off = 16; off; off >>= 1) tsum += __shfl_xor(tsum, off, 32);
        l = l * corr + tsum;
        acc0 *= corr; acc1 *= corr;
        for (int j = 0; j < 32; j++) {
            float pj = __shfl(p, j, 32);
            const float* vr = qkv + (size_t)(b * SEQ + s0 + j) * (3 * EMB) + 2 * EMB + h * HDIM;
            acc0 += pj * vr[lane];
            acc1 += pj * vr[lane + 32];
        }
        m = mnew;
    }
    size_t ro = (size_t)(b * SEQ + q) * EMB + h * HDIM;
    float inv = 1.0f / l;
    y[ro + lane]      = acc0 * inv;
    y[ro + lane + 32] = acc1 * inv;
}

// ---------------- per-row cross-entropy (logsumexp - target logit) ----------
__global__ __launch_bounds__(256)
void loss_row_kernel(const float* __restrict__ logits, const int* __restrict__ targets,
                     float* __restrict__ rowloss) {
    __shared__ float sm[256];
    int row = blockIdx.x, tid = threadIdx.x;
    const float* lr = logits + (size_t)row * NV;
    float mx = -1e30f;
    for (int c = tid; c < NV; c += 256) mx = fmaxf(mx, lr[c]);
    sm[tid] = mx; __syncthreads();
    for (int off = 128; off; off >>= 1) {
        if (tid < off) sm[tid] = fmaxf(sm[tid], sm[tid + off]);
        __syncthreads();
    }
    mx = sm[0]; __syncthreads();
    float se = 0.f;
    for (int c = tid; c < NV; c += 256) se += __expf(lr[c] - mx);
    sm[tid] = se; __syncthreads();
    for (int off = 128; off; off >>= 1) {
        if (tid < off) sm[tid] += sm[tid + off];
        __syncthreads();
    }
    if (tid == 0) {
        float lse = mx + __logf(sm[0]);
        rowloss[row] = lse - lr[targets[row]];
    }
}

__global__ __launch_bounds__(256)
void loss_reduce_kernel(const float* __restrict__ rowloss, float* __restrict__ out) {
    __shared__ float sm[256];
    int tid = threadIdx.x;
    float a = 0.f;
    for (int i = tid; i < NROWS; i += 256) a += rowloss[i];
    sm[tid] = a; __syncthreads();
    for (int off = 128; off; off >>= 1) {
        if (tid < off) sm[tid] += sm[tid + off];
        __syncthreads();
    }
    if (tid == 0) out[0] = sm[0] * (1.0f / (float)NROWS);
}

// ---------------- host-side orchestration ----------------
extern "C" void kernel_launch(void* const* d_in, const int* in_sizes, int n_in,
                              void* d_out, int out_size, void* d_ws, size_t ws_size,
                              hipStream_t stream) {
    (void)in_sizes; (void)n_in; (void)out_size; (void)ws_size;
    const int*   x0      = (const int*)d_in[0];
    const int*   targets = (const int*)d_in[1];
    const float* wte     = (const float*)d_in[2];
    const float* wpe     = (const float*)d_in[3];
    const float* ln1_w   = (const float*)d_in[4];
    const float* ln1_b   = (const float*)d_in[5];
    const float* Wqkv    = (const float*)d_in[6];
    const float* Wo      = (const float*)d_in[7];
    const float* ln2_w   = (const float*)d_in[8];
    const float* ln2_b   = (const float*)d_in[9];
    const float* Wfc     = (const float*)d_in[10];
    const float* Wproj   = (const float*)d_in[11];
    const float* lnf_w   = (const float*)d_in[12];
    const float* lnf_b   = (const float*)d_in[13];

    // carve workspace (~76 MB total)
    char* ws = (char*)d_ws;
    auto carve = [&](size_t bytes) -> char* {
        char* p = ws; ws += (bytes + 255) & ~(size_t)255; return p;
    };
    float*  xres = (float*) carve((size_t)NROWS * EMB * 4);        // residual stream
    __bf16* hbf  = (__bf16*)carve((size_t)NROWS * EMB * 2);        // LN output (GEMM A)
    float*  qkv  = (float*) carve((size_t)NROWS * 3 * EMB * 4);
    float*  ybuf = (float*) carve((size_t)NROWS * EMB * 4);        // attention out
    __bf16* ybf  = (__bf16*)carve((size_t)NROWS * EMB * 2);
    float*  fcb  = (float*) carve((size_t)NROWS * 4 * EMB * 4);    // gelu(h@Wfc)
    __bf16* fcbf = (__bf16*)carve((size_t)NROWS * 4 * EMB * 2);
    float*  rlss = (float*) carve((size_t)NROWS * 4);

    float* logits = (float*)d_out;
    float* lossp  = logits + (size_t)NROWS * NV;

    const dim3 blkG(128);
    const int  mTiles = NROWS / 128;   // 16

    embed_kernel<<<NROWS, 256, 0, stream>>>(x0, wte, wpe, xres);

    for (int l = 0; l < NL; l++) {
        // LN1 -> bf16
        ln_kernel<<<NROWS, 256, 0, stream>>>(xres, ln1_w + l * EMB, ln1_b + l * EMB, hbf);
        // qkv = h @ Wqkv[l]     [2048 x 2304]
        gemm_bf16_kernel<<<dim3((3 * EMB) / 64, mTiles), blkG, 0, stream>>>(
            hbf, Wqkv + (size_t)l * EMB * 3 * EMB, qkv, nullptr, 3 * EMB, EMB, 0, 0);
        // causal attention
        attn_kernel<<<(NBATCH * NH * SEQ) / 8, 256, 0, stream>>>(qkv, ybuf);
        cvt_bf16_kernel<<<(NROWS * EMB / 4 + 255) / 256, 256, 0, stream>>>(ybuf, ybf, NROWS * EMB / 4);
        // x = x + y @ Wo[l]
        gemm_bf16_kernel<<<dim3(EMB / 64, mTiles), blkG, 0, stream>>>(
            ybf, Wo + (size_t)l * EMB * EMB, xres, xres, EMB, EMB, 0, 0);
        // LN2 -> bf16
        ln_kernel<<<NROWS, 256, 0, stream>>>(xres, ln2_w + l * EMB, ln2_b + l * EMB, hbf);
        // fc = gelu(h @ Wfc[l])   [2048 x 3072]
        gemm_bf16_kernel<<<dim3((4 * EMB) / 64, mTiles), blkG, 0, stream>>>(
            hbf, Wfc + (size_t)l * EMB * 4 * EMB, fcb, nullptr, 4 * EMB, EMB, 0, 1);
        cvt_bf16_kernel<<<(NROWS * 4 * EMB / 4 + 255) / 256, 256, 0, stream>>>(fcb, fcbf, NROWS * 4 * EMB / 4);
        // x = x + fc @ Wproj[l]
        gemm_bf16_kernel<<<dim3(EMB / 64, mTiles), blkG, 0, stream>>>(
            fcbf, Wproj + (size_t)l * 4 * EMB * EMB, xres, xres, EMB, 4 * EMB, 0, 0);
    }

    // final LN + tied logits GEMM (B = wte accessed transposed), N edge guarded
    ln_kernel<<<NROWS, 256, 0, stream>>>(xres, lnf_w, lnf_b, hbf);
    gemm_bf16_kernel<<<dim3((NV + 63) / 64, mTiles), blkG, 0, stream>>>(
        hbf, wte, logits, nullptr, NV, EMB, 1, 0);

    // cross-entropy
    loss_row_kernel<<<NROWS, 256, 0, stream>>>(logits, targets, rlss);
    loss_reduce_kernel<<<1, 256, 0, stream>>>(rlss, lossp);
}
